// OptimalDiffusionClassifier_1451698946344
// MI455X (gfx1250) — matrix operand order repaired and checked by reading, not compiled, not run on hardware
//
#include <hip/hip_runtime.h>
#include <hip/hip_bf16.h>

typedef __attribute__((ext_vector_type(16))) __bf16 v16bf;
typedef __attribute__((ext_vector_type(8)))  float  v8f;

#define D_DIM   3072
#define N_SUP   500
#define N_PAD   512
#define T_STEPS 1000
#define B_Q     4
#define M_ROWS  (B_Q * T_STEPS)   // 4000
#define KTILE   32                 // 16 K-pairs per chunk
#define KPAIRS  16
#define MTILE   64
#define NTILE   128
#define LDS_W   20                 // padded stride (dwords): 16B-aligned b128, conflict-free
#define GEMM_THREADS 128           // 4 waves; each wave: 4 m-slots x 2 n-slots of C tiles

// ---------------------------------------------------------------------------
// Kernel 1: diffusion schedule (serial cumprod, T=1000, trivial one-shot)
// ---------------------------------------------------------------------------
__global__ void sched_kernel(float* sab, float* ab, float* invsig, float* coef) {
    if (blockIdx.x == 0 && threadIdx.x == 0) {
        float abp = 1.0f;
        for (int t = 0; t < T_STEPS; ++t) {
            float beta = 1e-4f + (0.02f - 1e-4f) * (float)t / (float)(T_STEPS - 1);
            abp *= (1.0f - beta);
            float s2 = 1.0f - abp;
            float sg = sqrtf(s2);
            sab[t]    = sqrtf(abp);
            ab[t]     = abp;
            invsig[t] = 1.0f / sg;
            coef[t]   = abp / s2;
        }
    }
}

// ---------------------------------------------------------------------------
// Kernel 2: dsq[b,n] = ||x_b - xs_n||^2  and  xssq[n] = ||xs_n||^2
// ---------------------------------------------------------------------------
__global__ void dsq_kernel(const float* __restrict__ x, const float* __restrict__ xs,
                           float* __restrict__ dsq, float* __restrict__ xssq) {
    __shared__ float red[256];
    const int tid = threadIdx.x;
    const int id  = blockIdx.x;
    float s = 0.0f;
    if (id < B_Q * N_SUP) {
        const int b = id / N_SUP, n = id % N_SUP;
        const float* xb  = x  + (size_t)b * D_DIM;
        const float* xsn = xs + (size_t)n * D_DIM;
        for (int d = tid; d < D_DIM; d += 256) {
            float df = xb[d] - xsn[d];
            s += df * df;
        }
    } else {
        const int n = id - B_Q * N_SUP;
        const float* xsn = xs + (size_t)n * D_DIM;
        for (int d = tid; d < D_DIM; d += 256) {
            float v = xsn[d];
            s += v * v;
        }
    }
    red[tid] = s;
    __syncthreads();
    for (int off = 128; off > 0; off >>= 1) {
        if (tid < off) red[tid] += red[tid + off];
        __syncthreads();
    }
    if (tid == 0) {
        if (id < B_Q * N_SUP) dsq[(id / N_SUP) * N_PAD + (id % N_SUP)] = red[0];
        else                  xssq[id - B_Q * N_SUP] = red[0];
    }
}

// ---------------------------------------------------------------------------
// Kernel 3: E[m,n] = eps_m . xs_n via bf16x3-split WMMA.
// hi/lo planes split once at LDS-fill time, stored in fragment word order ->
// fragment loads are pure aligned ds_load_b128. Each wave computes a 2x4 tile
// grid (24 WMMA : 24 b128 loads per chunk, 1:1 compute/LDS density).
// ---------------------------------------------------------------------------
union FragBF { v16bf v; uint4 q[2]; };

__device__ __forceinline__ void pack_pair(float f0, float f1,
                                          unsigned int& hiw, unsigned int& low) {
    unsigned int u0 = __float_as_uint(f0), u1 = __float_as_uint(f1);
    unsigned int h0 = u0 >> 16, h1 = u1 >> 16;            // truncated bf16 hi
    float r0 = f0 - __uint_as_float(h0 << 16);
    float r1 = f1 - __uint_as_float(h1 << 16);
    unsigned int l0 = __float_as_uint(r0) >> 16;
    unsigned int l1 = __float_as_uint(r1) >> 16;
    hiw = h0 | (h1 << 16);
    low = l0 | (l1 << 16);
}

__global__ void __launch_bounds__(GEMM_THREADS)
gemm_kernel(const float* __restrict__ eps, const float* __restrict__ xs,
            float* __restrict__ E) {
    __shared__ __align__(16) unsigned int sAhi[MTILE * LDS_W];
    __shared__ __align__(16) unsigned int sAlo[MTILE * LDS_W];
    __shared__ __align__(16) unsigned int sBhi[NTILE * LDS_W];
    __shared__ __align__(16) unsigned int sBlo[NTILE * LDS_W];

    const int tid   = threadIdx.x;
    const int lane  = tid & 31;
    const int wv    = tid >> 5;                 // 4 waves: wave w owns 32-col slot w
    const int half  = lane >> 4;                // K-half selector for fragments
    const int mbase = blockIdx.x * MTILE;
    const int nbase = blockIdx.y * NTILE;

    const v8f zero8 = {0.f, 0.f, 0.f, 0.f, 0.f, 0.f, 0.f, 0.f};
    v8f acc[4][2];
    for (int ms = 0; ms < 4; ++ms)
        for (int ns = 0; ns < 2; ++ns) acc[ms][ns] = zero8;

    #pragma unroll 2
    for (int k0 = 0; k0 < D_DIM; k0 += KTILE) {
        __syncthreads();
        // A fill: 64 rows x 16 K-pairs (float2 global loads)
        for (int e = tid; e < MTILE * KPAIRS; e += GEMM_THREADS) {
            int r = e >> 4, kp = e & 15;
            int row = mbase + r;
            float2 f = make_float2(0.f, 0.f);
            if (row < M_ROWS)
                f = ((const float2*)(eps + (size_t)row * D_DIM + k0))[kp];
            unsigned int hiw, low;
            pack_pair(f.x, f.y, hiw, low);
            sAhi[r * LDS_W + kp] = hiw;
            sAlo[r * LDS_W + kp] = low;
        }
        // B fill: 128 cols x 16 K-pairs
        for (int e = tid; e < NTILE * KPAIRS; e += GEMM_THREADS) {
            int n = e >> 4, kp = e & 15;
            int col = nbase + n;
            float2 f = make_float2(0.f, 0.f);
            if (col < N_SUP)
                f = ((const float2*)(xs + (size_t)col * D_DIM + k0))[kp];
            unsigned int hiw, low;
            pack_pair(f.x, f.y, hiw, low);
            sBhi[n * LDS_W + kp] = hiw;
            sBlo[n * LDS_W + kp] = low;
        }
        // prefetch next eps K-chunk (global_prefetch_b8)
        if (k0 + KTILE < D_DIM) {
            int row = mbase + ((tid * 2) & 63);
            if (row < M_ROWS)
                __builtin_prefetch(&eps[(size_t)row * D_DIM + k0 + KTILE], 0, 1);
        }
        __syncthreads();

        // B fragments (32x16 each): lanes 0-15 N=lane K=0..15, lanes 16-31 K=16..31
        FragBF bhi[2], blo[2];
        #pragma unroll
        for (int ns = 0; ns < 2; ++ns) {
            const int cb = (wv * 32 + ns * 16 + (lane & 15)) * LDS_W + half * 8;
            bhi[ns].q[0] = *(const uint4*)&sBhi[cb];
            bhi[ns].q[1] = *(const uint4*)&sBhi[cb + 4];
            blo[ns].q[0] = *(const uint4*)&sBlo[cb];
            blo[ns].q[1] = *(const uint4*)&sBlo[cb + 4];
        }

        #pragma unroll
        for (int ms = 0; ms < 4; ++ms) {
            // A fragment (16x32): lanes 0-15 M=lane K=0..15;
            //                     lanes 16-31 M=lane-16 K=16..31
            const int rb = (ms * 16 + (lane & 15)) * LDS_W + half * 8;
            FragBF ahi, alo;
            ahi.q[0] = *(const uint4*)&sAhi[rb];
            ahi.q[1] = *(const uint4*)&sAhi[rb + 4];
            alo.q[0] = *(const uint4*)&sAlo[rb];
            alo.q[1] = *(const uint4*)&sAlo[rb + 4];

            #pragma unroll
            for (int ns = 0; ns < 2; ++ns) {
                acc[ms][ns] = __builtin_amdgcn_wmma_f32_16x16x32_bf16(
                    false, ahi.v, false, bhi[ns].v, (short)0, acc[ms][ns], false, false);
                acc[ms][ns] = __builtin_amdgcn_wmma_f32_16x16x32_bf16(
                    false, ahi.v, false, blo[ns].v, (short)0, acc[ms][ns], false, false);
                acc[ms][ns] = __builtin_amdgcn_wmma_f32_16x16x32_bf16(
                    false, alo.v, false, bhi[ns].v, (short)0, acc[ms][ns], false, false);
            }
        }
    }

    // C store: vgpr v -> M = (lane<16 ? v : 8+v), N = lane&15
    const int nl = lane & 15;
    const int mh = half * 8;
    #pragma unroll
    for (int ms = 0; ms < 4; ++ms) {
        #pragma unroll
        for (int ns = 0; ns < 2; ++ns) {
            #pragma unroll
            for (int v = 0; v < 8; ++v) {
                int row = mbase + ms * 16 + mh + v;
                int col = nbase + wv * 32 + ns * 16 + nl;
                if (row < M_ROWS) E[(size_t)row * N_PAD + col] = acc[ms][ns][v];
            }
        }
    }
}

// ---------------------------------------------------------------------------
// Kernel 4: per-(b,t) row epilogue. xt_sq cancels inside the class softmax:
//   s_n = (sab_t * E - 0.5*ab_t*||xs_n||^2) / sigma_t
//   inner_c = (sum e^2 dsq) / (sum e)^2 ; partial[c,b,t] = inner_c * ab/sig^2
// One wave per row; lane l holds n = l + 32*j, j = 0..15.
// ---------------------------------------------------------------------------
__global__ void __launch_bounds__(256)
row_kernel(const float* __restrict__ E,
           const float* __restrict__ sab, const float* __restrict__ ab,
           const float* __restrict__ invsig, const float* __restrict__ coef,
           const float* __restrict__ xssq, const float* __restrict__ dsq,
           const int* __restrict__ ys, float* __restrict__ partial, int num_classes) {
    __shared__ int   ys_s[N_PAD];
    __shared__ float xssq_s[N_PAD];
    __shared__ float dsq_s[B_Q * N_PAD];

    const int tid = threadIdx.x;
    for (int i = tid; i < N_PAD; i += 256) {
        ys_s[i]   = (i < N_SUP) ? ys[i] : -1;
        xssq_s[i] = (i < N_SUP) ? xssq[i] : 0.0f;
    }
    for (int i = tid; i < B_Q * N_PAD; i += 256) dsq_s[i] = dsq[i];
    __syncthreads();

    const int lane = tid & 31;
    const int row  = blockIdx.x * 8 + (tid >> 5);     // (b,t) flattened
    const int b = row / T_STEPS, t = row % T_STEPS;

    const float sa = sab[t], a = ab[t], is = invsig[t];

    float sreg[16]; int creg[16]; float dreg[16];
    #pragma unroll
    for (int j = 0; j < 16; ++j) {
        int n = lane + 32 * j;
        float e = E[(size_t)row * N_PAD + n];
        creg[j] = ys_s[n];
        dreg[j] = dsq_s[b * N_PAD + n];
        sreg[j] = (n < N_SUP) ? (sa * e - 0.5f * a * xssq_s[n]) * is : 0.0f;
    }

    for (int c = 0; c < num_classes; ++c) {
        float mx = -3.0e38f;
        #pragma unroll
        for (int j = 0; j < 16; ++j)
            if (creg[j] == c) mx = fmaxf(mx, sreg[j]);
        for (int m = 16; m > 0; m >>= 1) mx = fmaxf(mx, __shfl_xor(mx, m, 32));

        float S = 0.0f, Q = 0.0f;
        #pragma unroll
        for (int j = 0; j < 16; ++j)
            if (creg[j] == c) {
                float e = __expf(sreg[j] - mx);
                S += e;
                Q += e * e * dreg[j];
            }
        for (int m = 16; m > 0; m >>= 1) {
            S += __shfl_xor(S, m, 32);
            Q += __shfl_xor(Q, m, 32);
        }
        if (lane == 0) {
            float inner = (S > 0.0f) ? Q / (S * S) : 0.0f;
            partial[((size_t)c * B_Q + b) * T_STEPS + t] = inner * coef[t];
        }
    }
}

// ---------------------------------------------------------------------------
// Kernel 5: out[b,c] = -(1/T) * sum_t partial[c,b,t]   (deterministic reduce)
// ---------------------------------------------------------------------------
__global__ void final_kernel(const float* __restrict__ partial, float* __restrict__ out,
                             int num_classes) {
    __shared__ float red[128];
    const int idx = blockIdx.x;            // idx = c*B_Q + b
    const int c = idx / B_Q, b = idx % B_Q;
    float s = 0.0f;
    for (int t = threadIdx.x; t < T_STEPS; t += 128)
        s += partial[(size_t)idx * T_STEPS + t];
    red[threadIdx.x] = s;
    __syncthreads();
    for (int off = 64; off > 0; off >>= 1) {
        if (threadIdx.x < off) red[threadIdx.x] += red[threadIdx.x + off];
        __syncthreads();
    }
    if (threadIdx.x == 0) out[b * num_classes + c] = -red[0] / (float)T_STEPS;
}

// ---------------------------------------------------------------------------
extern "C" void kernel_launch(void* const* d_in, const int* in_sizes, int n_in,
                              void* d_out, int out_size, void* d_ws, size_t ws_size,
                              hipStream_t stream) {
    (void)in_sizes; (void)n_in; (void)ws_size;
    const float* x   = (const float*)d_in[0];
    const float* xs  = (const float*)d_in[1];
    const float* eps = (const float*)d_in[2];
    const int*   ys  = (const int*)d_in[3];
    const int num_classes = out_size / B_Q;

    float* ws      = (float*)d_ws;
    float* sab     = ws + 0;
    float* ab      = ws + 1000;
    float* invsig  = ws + 2000;
    float* coef    = ws + 3000;
    float* xssq    = ws + 4096;
    float* dsq     = ws + 4608;                       // B_Q * N_PAD
    float* E       = ws + 8192;                       // M_ROWS * N_PAD
    float* partial = E + (size_t)M_ROWS * N_PAD;      // num_classes * B_Q * T_STEPS

    sched_kernel<<<1, 32, 0, stream>>>(sab, ab, invsig, coef);
    dsq_kernel<<<(B_Q + 1) * N_SUP, 256, 0, stream>>>(x, xs, dsq, xssq);

    dim3 gg((M_ROWS + MTILE - 1) / MTILE, N_PAD / NTILE);
    gemm_kernel<<<gg, GEMM_THREADS, 0, stream>>>(eps, xs, E);

    row_kernel<<<M_ROWS / 8, 256, 0, stream>>>(E, sab, ab, invsig, coef,
                                               xssq, dsq, ys, partial, num_classes);
    final_kernel<<<out_size, 128, 0, stream>>>(partial, (float*)d_out, num_classes);
}
